// GraphSAGEClassifier_9311489098206
// MI455X (gfx1250) — matrix operand order repaired and checked by reading, not compiled, not run on hardware
//
#include <hip/hip_runtime.h>

static constexpr int Nn  = 50000;
static constexpr int Ee  = 800000;
static constexpr int Din = 128;
static constexpr int Dh  = 256;
static constexpr int Dm  = 128;   // H/2
static constexpr int TB  = 256;
#define EPSV 1e-5f

typedef __attribute__((ext_vector_type(16))) __bf16 v16bf;
typedef __attribute__((ext_vector_type(8)))  __bf16 v8bf;
typedef __attribute__((ext_vector_type(4)))  __bf16 v4bf;
typedef __attribute__((ext_vector_type(8)))  float  v8f;

// ---------------- utility kernels ----------------

__global__ void zero_f32(float* __restrict__ p, int n) {
  int i = blockIdx.x * blockDim.x + threadIdx.x;
  if (i < n) p[i] = 0.0f;
}

__global__ void cvt_bf16(const float* __restrict__ x, __bf16* __restrict__ y, int n) {
  int i = blockIdx.x * blockDim.x + threadIdx.x;
  if (i < n) y[i] = (__bf16)x[i];
}

// wt[n*K + k] = w[k*Nout + n]   (w is K x Nout row-major; wt is Nout x K)
__global__ void cvt_wT(const float* __restrict__ w, __bf16* __restrict__ wt, int K, int Nout) {
  int i = blockIdx.x * blockDim.x + threadIdx.x;
  if (i >= K * Nout) return;
  int n = i / K;
  int k = i - n * K;
  wt[i] = (__bf16)w[(size_t)k * Nout + n];
}

__global__ void degree_kernel(const int* __restrict__ dst, float* __restrict__ deg, int E) {
  int e = blockIdx.x * blockDim.x + threadIdx.x;
  if (e < E) atomicAdd(&deg[dst[e]], 1.0f);
}

__global__ void invdeg_kernel(const float* __restrict__ deg, float* __restrict__ invd, int n) {
  int i = blockIdx.x * blockDim.x + threadIdx.x;
  if (i < n) invd[i] = 1.0f / fmaxf(deg[i], 1.0f);
}

// fold bias + batchnorm into per-feature affine: y = z*scale + shift
__global__ void make_affine(const float* __restrict__ bias, const float* __restrict__ gamma,
                            const float* __restrict__ beta, const float* __restrict__ mean,
                            const float* __restrict__ var, float* __restrict__ scale,
                            float* __restrict__ shift, int H) {
  int i = blockIdx.x * blockDim.x + threadIdx.x;
  if (i < H) {
    float s  = gamma[i] * rsqrtf(var[i] + EPSV);
    scale[i] = s;
    shift[i] = (bias[i] - mean[i]) * s + beta[i];
  }
}

// mean[i] = bf16(agg[i] * inv_deg[row])  -- one pass, K is a power of two
__global__ void mean_cvt(const float* __restrict__ agg, const float* __restrict__ invd,
                         __bf16* __restrict__ mn, int total, int shiftK) {
  int i = blockIdx.x * blockDim.x + threadIdx.x;
  if (i < total) mn[i] = (__bf16)(agg[i] * invd[i >> shiftK]);
}

// ---------------- edge scatter-sum ----------------
// one wave per edge; lane covers K/32 contiguous features
__global__ void aggregate_kernel(const __bf16* __restrict__ h, const int* __restrict__ src,
                                 const int* __restrict__ dst, float* __restrict__ agg,
                                 int K, int E) {
  int wid = blockIdx.x * (blockDim.x >> 5) + (threadIdx.x >> 5);
  if (wid >= E) return;
  int lane = threadIdx.x & 31;
  int s = src[wid];
  int d = dst[wid];
  if (K == Dh) {
    const v8bf v = *(const v8bf*)(h + (size_t)s * Dh + lane * 8);
    float* ap = agg + (size_t)d * Dh + lane * 8;
#pragma unroll
    for (int j = 0; j < 8; ++j) atomicAdd(ap + j, (float)v[j]);
  } else {
    const v4bf v = *(const v4bf*)(h + (size_t)s * Din + lane * 4);
    float* ap = agg + (size_t)d * Din + lane * 4;
#pragma unroll
    for (int j = 0; j < 4; ++j) atomicAdd(ap + j, (float)v[j]);
  }
}

// A-fragment loader (contiguous k order: lane half kg holds k = kb + 16*kg .. +15, matching B)
__device__ inline v16bf load_afrag(const __bf16* __restrict__ row, int kb, int kg) {
  v16bf a = *(const v16bf*)(row + kb + kg * 16);
  return a;
}

// ---------------- fused SAGE GEMM, 16x64 tile per wave ----------------
// out = relu( (aM@Wn + h@Wr) * scale + shift )       (aM != nullptr)
// out = relu( (h@Wn) * scale + shift )               (aM == nullptr; scale==nullptr -> 1)
__global__ __launch_bounds__(TB)
void sage_gemm4(const __bf16* __restrict__ aM, const __bf16* __restrict__ h,
                const __bf16* __restrict__ wtn, const __bf16* __restrict__ wtr,
                const float* __restrict__ scale, const float* __restrict__ shift,
                __bf16* __restrict__ out, int K, int Hout, int rowTiles) {
  const int wid = blockIdx.x * (blockDim.x >> 5) + (threadIdx.x >> 5);
  const int nct = Hout >> 6;            // 64-wide column supertiles
  if (wid >= rowTiles * nct) return;
  const int rt = wid / nct;
  const int ct = wid - rt * nct;
  const int m0 = rt << 4;
  const int n0 = ct << 6;
  const int lane = threadIdx.x & 31;
  const int lm = lane & 15;
  const int kg = lane >> 4;

  const __bf16* hrow = h + (size_t)(m0 + lm) * K;
  const size_t  tstr = (size_t)16 * K;
  const __bf16* bnp0 = wtn + (size_t)(n0 + lm) * K;
  const __bf16* bnp1 = bnp0 + tstr;
  const __bf16* bnp2 = bnp0 + 2 * tstr;
  const __bf16* bnp3 = bnp0 + 3 * tstr;

  v8f acc[4];
#pragma unroll
  for (int t = 0; t < 4; ++t) acc[t] = v8f{0.f, 0.f, 0.f, 0.f, 0.f, 0.f, 0.f, 0.f};

  if (aM) {
    const __bf16* mrow = aM + (size_t)(m0 + lm) * K;
    const __bf16* brp0 = wtr + (size_t)(n0 + lm) * K;
    const __bf16* brp1 = brp0 + tstr;
    const __bf16* brp2 = brp0 + 2 * tstr;
    const __bf16* brp3 = brp0 + 3 * tstr;
    for (int kb = 0; kb < K; kb += 32) {
      const int bo = kb + kg * 16;
      // issue all fragment loads first, then the 8-WMMA burst
      const v16bf aMn = load_afrag(mrow, kb, kg);
      const v16bf aH  = load_afrag(hrow, kb, kg);
      const v16bf bN0 = *(const v16bf*)(bnp0 + bo);
      const v16bf bN1 = *(const v16bf*)(bnp1 + bo);
      const v16bf bN2 = *(const v16bf*)(bnp2 + bo);
      const v16bf bN3 = *(const v16bf*)(bnp3 + bo);
      const v16bf bR0 = *(const v16bf*)(brp0 + bo);
      const v16bf bR1 = *(const v16bf*)(brp1 + bo);
      const v16bf bR2 = *(const v16bf*)(brp2 + bo);
      const v16bf bR3 = *(const v16bf*)(brp3 + bo);
      acc[0] = __builtin_amdgcn_wmma_f32_16x16x32_bf16(false, aMn, false, bN0, (short)0, acc[0], false, false);
      acc[1] = __builtin_amdgcn_wmma_f32_16x16x32_bf16(false, aMn, false, bN1, (short)0, acc[1], false, false);
      acc[2] = __builtin_amdgcn_wmma_f32_16x16x32_bf16(false, aMn, false, bN2, (short)0, acc[2], false, false);
      acc[3] = __builtin_amdgcn_wmma_f32_16x16x32_bf16(false, aMn, false, bN3, (short)0, acc[3], false, false);
      acc[0] = __builtin_amdgcn_wmma_f32_16x16x32_bf16(false, aH,  false, bR0, (short)0, acc[0], false, false);
      acc[1] = __builtin_amdgcn_wmma_f32_16x16x32_bf16(false, aH,  false, bR1, (short)0, acc[1], false, false);
      acc[2] = __builtin_amdgcn_wmma_f32_16x16x32_bf16(false, aH,  false, bR2, (short)0, acc[2], false, false);
      acc[3] = __builtin_amdgcn_wmma_f32_16x16x32_bf16(false, aH,  false, bR3, (short)0, acc[3], false, false);
    }
  } else {
    for (int kb = 0; kb < K; kb += 32) {
      const int bo = kb + kg * 16;
      const v16bf aH  = load_afrag(hrow, kb, kg);
      const v16bf bN0 = *(const v16bf*)(bnp0 + bo);
      const v16bf bN1 = *(const v16bf*)(bnp1 + bo);
      const v16bf bN2 = *(const v16bf*)(bnp2 + bo);
      const v16bf bN3 = *(const v16bf*)(bnp3 + bo);
      acc[0] = __builtin_amdgcn_wmma_f32_16x16x32_bf16(false, aH, false, bN0, (short)0, acc[0], false, false);
      acc[1] = __builtin_amdgcn_wmma_f32_16x16x32_bf16(false, aH, false, bN1, (short)0, acc[1], false, false);
      acc[2] = __builtin_amdgcn_wmma_f32_16x16x32_bf16(false, aH, false, bN2, (short)0, acc[2], false, false);
      acc[3] = __builtin_amdgcn_wmma_f32_16x16x32_bf16(false, aH, false, bN3, (short)0, acc[3], false, false);
    }
  }

  // epilogue: per-feature affine + ReLU, store bf16.
  // D layout: VGPR r holds row m0 + kg*8 + r, column nIdx (ISA 32-bit C/D 16x16 layout)
  const int rb = m0 + kg * 8;
#pragma unroll
  for (int t = 0; t < 4; ++t) {
    const int nIdx = n0 + (t << 4) + lm;
    const float sv = scale ? scale[nIdx] : 1.0f;
    const float tv = shift[nIdx];
#pragma unroll
    for (int r = 0; r < 8; ++r) {
      float v = fmaf(acc[t][r], sv, tv);
      v = fmaxf(v, 0.0f);
      out[(size_t)(rb + r) * Hout + nIdx] = (__bf16)v;
    }
  }
}

// ---------------- final 128->2 projection + log_softmax ----------------
// one wave per node, wave32 shuffle reduction
__global__ void classify_kernel(const __bf16* __restrict__ h4, const float* __restrict__ w2,
                                const float* __restrict__ b2, float* __restrict__ out, int n) {
  int wid = blockIdx.x * (blockDim.x >> 5) + (threadIdx.x >> 5);
  if (wid >= n) return;
  int lane = threadIdx.x & 31;
  const __bf16* hp = h4 + (size_t)wid * Dm + lane * 4;
  float p0 = 0.f, p1 = 0.f;
#pragma unroll
  for (int j = 0; j < 4; ++j) {
    float v = (float)hp[j];
    int k = lane * 4 + j;
    p0 = fmaf(v, w2[k * 2 + 0], p0);
    p1 = fmaf(v, w2[k * 2 + 1], p1);
  }
#pragma unroll
  for (int off = 16; off > 0; off >>= 1) {
    p0 += __shfl_xor(p0, off, 32);
    p1 += __shfl_xor(p1, off, 32);
  }
  if (lane == 0) {
    float z0 = p0 + b2[0];
    float z1 = p1 + b2[1];
    float m = fmaxf(z0, z1);
    float lse = m + logf(__expf(z0 - m) + __expf(z1 - m));
    out[wid * 2 + 0] = z0 - lse;
    out[wid * 2 + 1] = z1 - lse;
  }
}

// ---------------- host orchestration ----------------

extern "C" void kernel_launch(void* const* d_in, const int* in_sizes, int n_in,
                              void* d_out, int out_size, void* d_ws, size_t ws_size,
                              hipStream_t stream) {
  const float* x       = (const float*)d_in[0];
  const int*   ei      = (const int*)  d_in[1];
  const float* w_nei0  = (const float*)d_in[2];
  const float* w_root0 = (const float*)d_in[3];
  const float* b0      = (const float*)d_in[4];
  const float* w_nei   = (const float*)d_in[5];
  const float* w_root  = (const float*)d_in[6];
  const float* bvec    = (const float*)d_in[7];
  const float* gamma   = (const float*)d_in[8];
  const float* beta    = (const float*)d_in[9];
  const float* mean    = (const float*)d_in[10];
  const float* var     = (const float*)d_in[11];
  const float* w1      = (const float*)d_in[12];
  const float* b1      = (const float*)d_in[13];
  const float* w2      = (const float*)d_in[14];
  const float* b2      = (const float*)d_in[15];
  float* out = (float*)d_out;

  const int* src = ei;
  const int* dst = ei + Ee;

  // workspace carve (aligned to 256B)
  char* wp = (char*)d_ws;
  auto carve = [&](size_t bytes) -> char* {
    char* r = wp;
    wp += (bytes + 255) & ~(size_t)255;
    return r;
  };
  __bf16* hA    = (__bf16*)carve((size_t)Nn * Dh * 2);
  __bf16* hB    = (__bf16*)carve((size_t)Nn * Dh * 2);
  __bf16* hM    = (__bf16*)carve((size_t)Nn * Dh * 2);   // bf16 neighbor-mean matrix
  float*  agg   = (float*) carve((size_t)Nn * Dh * 4);
  float*  deg   = (float*) carve((size_t)Nn * 4);
  float*  invd  = (float*) carve((size_t)Nn * 4);
  float*  scale = (float*) carve(Dh * 4);
  float*  shift = (float*) carve(Dh * 4);
  __bf16* wtn0  = (__bf16*)carve((size_t)Dh * Din * 2);
  __bf16* wtr0  = (__bf16*)carve((size_t)Dh * Din * 2);
  __bf16* wtn1  = (__bf16*)carve((size_t)Dh * Dh * 2);
  __bf16* wtr1  = (__bf16*)carve((size_t)Dh * Dh * 2);
  __bf16* wtn2  = (__bf16*)carve((size_t)Dh * Dh * 2);
  __bf16* wtr2  = (__bf16*)carve((size_t)Dh * Dh * 2);
  __bf16* wt1   = (__bf16*)carve((size_t)Dm * Dh * 2);
  __bf16* h4    = (__bf16*)agg;  // agg is dead by the time h4 is written

  auto cdiv = [](int a, int b) { return (a + b - 1) / b; };
  const int rowTiles  = Nn / 16;                 // 3125
  const int aggBlocks = cdiv(Ee, TB / 32);       // wave per edge
  const int clsBlocks = cdiv(Nn, TB / 32);       // wave per node

  // input + weight conversion to bf16 (weights transposed to [n][k])
  cvt_bf16<<<cdiv(Nn * Din, TB), TB, 0, stream>>>(x, hA, Nn * Din);
  cvt_wT<<<cdiv(Din * Dh, TB), TB, 0, stream>>>(w_nei0,  wtn0, Din, Dh);
  cvt_wT<<<cdiv(Din * Dh, TB), TB, 0, stream>>>(w_root0, wtr0, Din, Dh);
  cvt_wT<<<cdiv(Dh * Dh, TB), TB, 0, stream>>>(w_nei,            wtn1, Dh, Dh);
  cvt_wT<<<cdiv(Dh * Dh, TB), TB, 0, stream>>>(w_nei + Dh * Dh,  wtn2, Dh, Dh);
  cvt_wT<<<cdiv(Dh * Dh, TB), TB, 0, stream>>>(w_root,           wtr1, Dh, Dh);
  cvt_wT<<<cdiv(Dh * Dh, TB), TB, 0, stream>>>(w_root + Dh * Dh, wtr2, Dh, Dh);
  cvt_wT<<<cdiv(Dh * Dm, TB), TB, 0, stream>>>(w1, wt1, Dh, Dm);

  // degrees
  zero_f32<<<cdiv(Nn, TB), TB, 0, stream>>>(deg, Nn);
  degree_kernel<<<cdiv(Ee, TB), TB, 0, stream>>>(dst, deg, Ee);
  invdeg_kernel<<<cdiv(Nn, TB), TB, 0, stream>>>(deg, invd, Nn);

  // ---- layer 0: 128 -> 256, hA -> hB ----
  zero_f32<<<cdiv(Nn * Din, TB), TB, 0, stream>>>(agg, Nn * Din);
  aggregate_kernel<<<aggBlocks, TB, 0, stream>>>(hA, src, dst, agg, Din, Ee);
  mean_cvt<<<cdiv(Nn * Din, TB), TB, 0, stream>>>(agg, invd, hM, Nn * Din, 7);
  make_affine<<<1, Dh, 0, stream>>>(b0, gamma, beta, mean, var, scale, shift, Dh);
  {
    int waves = rowTiles * (Dh / 64);
    sage_gemm4<<<cdiv(waves, TB / 32), TB, 0, stream>>>(hM, hA, wtn0, wtr0,
                                                        scale, shift, hB, Din, Dh, rowTiles);
  }

  // ---- layer 1: 256 -> 256, hB -> hA ----
  zero_f32<<<cdiv(Nn * Dh, TB), TB, 0, stream>>>(agg, Nn * Dh);
  aggregate_kernel<<<aggBlocks, TB, 0, stream>>>(hB, src, dst, agg, Dh, Ee);
  mean_cvt<<<cdiv(Nn * Dh, TB), TB, 0, stream>>>(agg, invd, hM, Nn * Dh, 8);
  make_affine<<<1, Dh, 0, stream>>>(bvec, gamma + Dh, beta + Dh, mean + Dh, var + Dh,
                                    scale, shift, Dh);
  {
    int waves = rowTiles * (Dh / 64);
    sage_gemm4<<<cdiv(waves, TB / 32), TB, 0, stream>>>(hM, hB, wtn1, wtr1,
                                                        scale, shift, hA, Dh, Dh, rowTiles);
  }

  // ---- layer 2: 256 -> 256, hA -> hB ----
  zero_f32<<<cdiv(Nn * Dh, TB), TB, 0, stream>>>(agg, Nn * Dh);
  aggregate_kernel<<<aggBlocks, TB, 0, stream>>>(hA, src, dst, agg, Dh, Ee);
  mean_cvt<<<cdiv(Nn * Dh, TB), TB, 0, stream>>>(agg, invd, hM, Nn * Dh, 8);
  make_affine<<<1, Dh, 0, stream>>>(bvec + Dh, gamma + 2 * Dh, beta + 2 * Dh,
                                    mean + 2 * Dh, var + 2 * Dh, scale, shift, Dh);
  {
    int waves = rowTiles * (Dh / 64);
    sage_gemm4<<<cdiv(waves, TB / 32), TB, 0, stream>>>(hM, hA, wtn2, wtr2,
                                                        scale, shift, hB, Dh, Dh, rowTiles);
  }

  // ---- MLP: h4 = relu(hB @ w1 + b1)  (256 -> 128; agg is dead, reuse as h4) ----
  {
    int waves = rowTiles * (Dm / 64);
    sage_gemm4<<<cdiv(waves, TB / 32), TB, 0, stream>>>(nullptr, hB, wt1, nullptr,
                                                        nullptr, b1, h4, Dh, Dm, rowTiles);
  }

  // ---- logits + log_softmax ----
  classify_kernel<<<clsBlocks, TB, 0, stream>>>(h4, w2, b2, out, Nn);
}
